// RGAT_73254962201302
// MI455X (gfx1250) — matrix operand order repaired and checked by reading, not compiled, not run on hardware
//
#include <hip/hip_runtime.h>
#include <hip/hip_bf16.h>

#define NN   20000
#define EE   320000
#define RR   3
#define LL   3
#define HH   4
#define DD   64
#define INS  256
#define HD   256   // H*D
#define OUTS 128
#define NEG_SLOPE 0.2f

typedef __attribute__((ext_vector_type(16))) __bf16 v16bf;
typedef __attribute__((ext_vector_type(8)))  float  v8f;

// ---------------------------------------------------------------------------
// Pack a row-major f32 weight matrix [K=256][Nc] into bf16 WMMA B-fragments.
// Per 32(K)x16(N) tile: element t = tile*512 + lane*16 + j holds
//   W[kt*32 + (lane/16)*16 + j][nt*16 + lane%16]
// so the GEMM can load each lane's 16 bf16 contiguously (two b128 loads).
// ---------------------------------------------------------------------------
__global__ void pack_w_kernel(const float* __restrict__ Wsrc,
                              __bf16* __restrict__ Wdst, int Nc) {
    int t = blockIdx.x * blockDim.x + threadIdx.x;
    int total = 256 * Nc;
    if (t >= total) return;
    int tile   = t >> 9;        // /512
    int within = t & 511;
    int lane   = within >> 4;
    int j      = within & 15;
    int ntiles = Nc >> 4;
    int kt = tile / ntiles, nt = tile % ntiles;
    int k  = kt * 32 + ((lane >> 4) << 4) + j;
    int n  = nt * 16 + (lane & 15);
    Wdst[t] = (__bf16)Wsrc[k * Nc + n];
}

// ---------------------------------------------------------------------------
// Convert f32 activations (+ optional per-column 3-way bias sum, optional
// ReLU) to bf16 row-major for WMMA A operand.
// ---------------------------------------------------------------------------
__global__ void cvt_bf16_kernel(const float* __restrict__ src,
                                const float* __restrict__ bias3, // [3][256] or null
                                __bf16* __restrict__ dst,
                                int total, int do_relu) {
    int t = blockIdx.x * blockDim.x + threadIdx.x;
    if (t >= total) return;
    float v = src[t];
    if (bias3) {
        int c = t & (HD - 1);
        v += bias3[c] + bias3[HD + c] + bias3[2 * HD + c];
    }
    if (do_relu && v < 0.f) v = 0.f;
    dst[t] = (__bf16)v;
}

// ---------------------------------------------------------------------------
// C[M,Nc] = A_bf16[M,256] x Bpacked(256,Nc)  (+ optional bias[Nc]), f32 out.
// One wave computes a 16x64 output strip (4 N-tiles sharing one A fragment):
// per K-step of 32 -> 2 A loads + 8 B loads + 4 v_wmma_f32_16x16x32_bf16.
// 8 waves / block. Nc must be a multiple of 64 (256 and 128 both are).
// ---------------------------------------------------------------------------
__global__ void gemm_bf16_wmma_kernel(const __bf16* __restrict__ A,
                                      const __bf16* __restrict__ Bp,
                                      float* __restrict__ C,
                                      const float* __restrict__ biasv,
                                      int M, int Nc) {
    const int K = 256;
    int wave = (blockIdx.x * blockDim.x + threadIdx.x) >> 5;
    int lane = threadIdx.x & 31;
    int ntiles  = Nc >> 4;         // 16-wide tiles
    int nstrips = Nc >> 6;         // 64-wide strips (4 tiles)
    int mtiles  = M >> 4;
    if (wave >= mtiles * nstrips) return;
    int mt = wave / nstrips, ns = wave % nstrips;
    int m0  = mt << 4;
    int nt0 = ns << 2;             // first 16-wide tile of this strip

    int mrow  = m0 + (lane & 15);
    int khalf = (lane >> 4) << 4;          // 0 for lanes 0-15, 16 for 16-31
    const __bf16* arow = A + (size_t)mrow * K;

    v8f c0 = {0.f,0.f,0.f,0.f,0.f,0.f,0.f,0.f};
    v8f c1 = c0, c2 = c0, c3 = c0;

    #pragma unroll
    for (int kt = 0; kt < K / 32; ++kt) {
        int k0 = kt << 5;
        union { v16bf v; uint4 q[2]; } a, b0, b1, b2, b3;
        // A fragment: lanes 0-15 need K = k0..k0+15,
        //             lanes 16-31 need K = k0+16..k0+31.
        a.q[0] = *(const uint4*)(arow + k0 + khalf);
        a.q[1] = *(const uint4*)(arow + k0 + khalf + 8);
        // B fragments: pre-packed contiguous 16 bf16 per lane per tile.
        const __bf16* bt = Bp + (((size_t)kt * ntiles + nt0) << 9) + (lane << 4);
        b0.q[0] = *(const uint4*)(bt);
        b0.q[1] = *(const uint4*)(bt + 8);
        b1.q[0] = *(const uint4*)(bt + 512);
        b1.q[1] = *(const uint4*)(bt + 512 + 8);
        b2.q[0] = *(const uint4*)(bt + 1024);
        b2.q[1] = *(const uint4*)(bt + 1024 + 8);
        b3.q[0] = *(const uint4*)(bt + 1536);
        b3.q[1] = *(const uint4*)(bt + 1536 + 8);
        c0 = __builtin_amdgcn_wmma_f32_16x16x32_bf16(
                 false, a.v, false, b0.v, (short)0, c0, false, false);
        c1 = __builtin_amdgcn_wmma_f32_16x16x32_bf16(
                 false, a.v, false, b1.v, (short)0, c1, false, false);
        c2 = __builtin_amdgcn_wmma_f32_16x16x32_bf16(
                 false, a.v, false, b2.v, (short)0, c2, false, false);
        c3 = __builtin_amdgcn_wmma_f32_16x16x32_bf16(
                 false, a.v, false, b3.v, (short)0, c3, false, false);
    }
    // C/D layout: VGPR v -> row m0 + v + 8*(lane>=16), col n0 + lane%16
    int crow0 = m0 + ((lane >> 4) << 3);
    int ccol  = (nt0 << 4) + (lane & 15);
    float* crow = C + (size_t)crow0 * Nc + ccol;
    float bb0 = biasv ? biasv[ccol]      : 0.f;
    float bb1 = biasv ? biasv[ccol + 16] : 0.f;
    float bb2 = biasv ? biasv[ccol + 32] : 0.f;
    float bb3 = biasv ? biasv[ccol + 48] : 0.f;
    #pragma unroll
    for (int v = 0; v < 8; ++v) {
        float* p = crow + (size_t)v * Nc;
        p[0]  = c0[v] + bb0;
        p[16] = c1[v] + bb1;
        p[32] = c2[v] + bb2;
        p[48] = c3[v] + bb3;
    }
}

// ---------------------------------------------------------------------------
// el[n,h] = <feat[n,h,:], attn_l[h,:]> ; er likewise. Thread per (n,h).
// ---------------------------------------------------------------------------
__global__ void elr_kernel(const float* __restrict__ feat,
                           const float* __restrict__ al,
                           const float* __restrict__ ar,
                           float* __restrict__ el, float* __restrict__ er) {
    int t = blockIdx.x * blockDim.x + threadIdx.x;
    if (t >= NN * HH) return;
    int n = t >> 2, h = t & 3;
    const float4* f = (const float4*)(feat + (size_t)n * HD + h * DD);
    const float4* a = (const float4*)(al + h * DD);
    const float4* b = (const float4*)(ar + h * DD);
    float sl = 0.f, sr = 0.f;
    #pragma unroll
    for (int i = 0; i < 16; ++i) {
        float4 fv = f[i], av = a[i], bv = b[i];
        sl += fv.x * av.x + fv.y * av.y + fv.z * av.z + fv.w * av.w;
        sr += fv.x * bv.x + fv.y * bv.y + fv.z * bv.z + fv.w * bv.w;
    }
    el[t] = sl; er[t] = sr;
}

__global__ void init_ms_kernel(float* __restrict__ m, float* __restrict__ s) {
    int t = blockIdx.x * blockDim.x + threadIdx.x;
    if (t >= NN * HH) return;
    m[t] = __int_as_float(0xFF800000); // -inf
    s[t] = 0.f;
}

__device__ __forceinline__ float leaky(float x) {
    return x >= 0.f ? x : NEG_SLOPE * x;
}
// Float atomic max via signed-int max for >=0 and unsigned-int min for <0.
__device__ __forceinline__ void atomicMaxF(float* addr, float val) {
    if (val >= 0.f) atomicMax((int*)addr, __float_as_int(val));
    else            atomicMin((unsigned int*)addr, __float_as_uint(val));
}

// segment max over incoming edges; thread per edge, 4 heads each.
__global__ void edge_max_kernel(const int* __restrict__ src,
                                const int* __restrict__ dst,
                                const float* __restrict__ el,
                                const float* __restrict__ er,
                                float* __restrict__ m) {
    int e = blockIdx.x * blockDim.x + threadIdx.x;
    if (e >= EE) return;
    int u = src[e], v = dst[e];
    float4 a = *(const float4*)(el + (size_t)u * 4);
    float4 b = *(const float4*)(er + (size_t)v * 4);
    atomicMaxF(&m[v * 4 + 0], leaky(a.x + b.x));
    atomicMaxF(&m[v * 4 + 1], leaky(a.y + b.y));
    atomicMaxF(&m[v * 4 + 2], leaky(a.z + b.z));
    atomicMaxF(&m[v * 4 + 3], leaky(a.w + b.w));
}

// segment sum of exp(e - m); thread per edge.
__global__ void edge_sum_kernel(const int* __restrict__ src,
                                const int* __restrict__ dst,
                                const float* __restrict__ el,
                                const float* __restrict__ er,
                                const float* __restrict__ m,
                                float* __restrict__ s) {
    int e = blockIdx.x * blockDim.x + threadIdx.x;
    if (e >= EE) return;
    int u = src[e], v = dst[e];
    float4 a  = *(const float4*)(el + (size_t)u * 4);
    float4 b  = *(const float4*)(er + (size_t)v * 4);
    float4 mv = *(const float4*)(m + (size_t)v * 4);
    atomicAdd(&s[v * 4 + 0], __expf(leaky(a.x + b.x) - mv.x));
    atomicAdd(&s[v * 4 + 1], __expf(leaky(a.y + b.y) - mv.y));
    atomicAdd(&s[v * 4 + 2], __expf(leaky(a.z + b.z) - mv.z));
    atomicAdd(&s[v * 4 + 3], __expf(leaky(a.w + b.w) - mv.w));
}

// acc[dst] += alpha * feat[src]; one wave per edge, 8 floats per lane
// (each lane's 8 elements sit inside a single head: head = lane/8).
__global__ void edge_scatter_kernel(const float* __restrict__ feat,
                                    const int* __restrict__ src,
                                    const int* __restrict__ dst,
                                    const float* __restrict__ el,
                                    const float* __restrict__ er,
                                    const float* __restrict__ m,
                                    const float* __restrict__ s,
                                    float* __restrict__ acc) {
    int gid = blockIdx.x * blockDim.x + threadIdx.x;
    int e = gid >> 5;
    if (e >= EE) return;
    int lane = gid & 31;
    int u = src[e], v = dst[e];
    int h = lane >> 3;
    float ev = leaky(el[u * 4 + h] + er[v * 4 + h]);
    float alpha = __expf(ev - m[v * 4 + h]) / s[v * 4 + h];
    const float* fsrc = feat + (size_t)u * HD + (lane << 3);
    float* adst = acc + (size_t)v * HD + (lane << 3);
    float4 f0 = *(const float4*)(fsrc);
    float4 f1 = *(const float4*)(fsrc + 4);
    atomicAdd(adst + 0, alpha * f0.x);
    atomicAdd(adst + 1, alpha * f0.y);
    atomicAdd(adst + 2, alpha * f0.z);
    atomicAdd(adst + 3, alpha * f0.w);
    atomicAdd(adst + 4, alpha * f1.x);
    atomicAdd(adst + 5, alpha * f1.y);
    atomicAdd(adst + 6, alpha * f1.z);
    atomicAdd(adst + 7, alpha * f1.w);
}

// ---------------------------------------------------------------------------
extern "C" void kernel_launch(void* const* d_in, const int* in_sizes, int n_in,
                              void* d_out, int out_size, void* d_ws, size_t ws_size,
                              hipStream_t stream) {
    const float* x        = (const float*)d_in[0];
    const int*   edge_src = (const int*)  d_in[1];
    const int*   edge_dst = (const int*)  d_in[2];
    const float* W        = (const float*)d_in[3];
    const float* attn_l   = (const float*)d_in[4];
    const float* attn_r   = (const float*)d_in[5];
    const float* bias     = (const float*)d_in[6];
    const float* fc_w     = (const float*)d_in[7];
    const float* fc_b     = (const float*)d_in[8];
    float* out = (float*)d_out;

    // ---- workspace carve-out (256B aligned) ----
    char*  base = (char*)d_ws;
    size_t off  = 0;
    auto carve = [&](size_t bytes) -> char* {
        char* p = base + off;
        off = (off + bytes + 255) & ~(size_t)255;
        return p;
    };
    const size_t WMAT = 256 * 256;                     // packed elems per rel matrix
    __bf16* Wp   = (__bf16*)carve((RR * LL * WMAT + 256 * 128) * sizeof(__bf16));
    __bf16* hb   = (__bf16*)carve((size_t)NN * HD * sizeof(__bf16));
    float*  feat = (float*) carve((size_t)NN * HD * sizeof(float));
    float*  acc  = (float*) carve((size_t)NN * HD * sizeof(float));
    float*  el   = (float*) carve((size_t)NN * HH * sizeof(float));
    float*  er   = (float*) carve((size_t)NN * HH * sizeof(float));
    float*  mbuf = (float*) carve((size_t)NN * HH * sizeof(float));
    float*  sbuf = (float*) carve((size_t)NN * HH * sizeof(float));
    (void)ws_size; (void)in_sizes; (void)n_in; (void)out_size;

    const int TB = 256;

    // ---- pack all weights into WMMA B-fragment order (bf16) ----
    for (int lr = 0; lr < LL * RR; ++lr)
        pack_w_kernel<<<(256 * 256 + TB - 1) / TB, TB, 0, stream>>>(
            W + (size_t)lr * INS * HD, Wp + (size_t)lr * WMAT, HD);
    __bf16* Wp_fc = Wp + (size_t)LL * RR * WMAT;
    pack_w_kernel<<<(256 * 128 + TB - 1) / TB, TB, 0, stream>>>(fc_w, Wp_fc, OUTS);

    // ---- layer 0 input: bf16(x) ----
    cvt_bf16_kernel<<<((size_t)NN * HD + TB - 1) / TB, TB, 0, stream>>>(
        x, nullptr, hb, NN * HD, 0);

    const int gemm_waves_feat  = (NN / 16) * (HD / 64);
    const int gemm_blocks_feat = (gemm_waves_feat * 32 + TB - 1) / TB;
    const int nh_blocks = (NN * HH + TB - 1) / TB;
    const int e_blocks  = (EE + TB - 1) / TB;
    const int sc_blocks = ((size_t)EE * 32 + TB - 1) / TB;

    for (int l = 0; l < LL; ++l) {
        hipMemsetAsync(acc, 0, (size_t)NN * HD * sizeof(float), stream);
        for (int r = 0; r < RR; ++r) {
            int lr = l * RR + r;
            // feat = h @ W[l,r]   (bf16 WMMA, f32 accumulate)
            gemm_bf16_wmma_kernel<<<gemm_blocks_feat, TB, 0, stream>>>(
                hb, Wp + (size_t)lr * WMAT, feat, nullptr, NN, HD);
            // attention logits per node
            elr_kernel<<<nh_blocks, TB, 0, stream>>>(
                feat, attn_l + (size_t)lr * HH * DD, attn_r + (size_t)lr * HH * DD,
                el, er);
            // segment softmax + weighted scatter
            init_ms_kernel<<<nh_blocks, TB, 0, stream>>>(mbuf, sbuf);
            edge_max_kernel<<<e_blocks, TB, 0, stream>>>(
                edge_src + (size_t)r * EE, edge_dst + (size_t)r * EE, el, er, mbuf);
            edge_sum_kernel<<<e_blocks, TB, 0, stream>>>(
                edge_src + (size_t)r * EE, edge_dst + (size_t)r * EE, el, er, mbuf, sbuf);
            edge_scatter_kernel<<<sc_blocks, TB, 0, stream>>>(
                feat, edge_src + (size_t)r * EE, edge_dst + (size_t)r * EE,
                el, er, mbuf, sbuf, acc);
        }
        // h_next = [relu](acc + sum_r bias[l,r])  -> bf16 for next GEMM
        cvt_bf16_kernel<<<((size_t)NN * HD + TB - 1) / TB, TB, 0, stream>>>(
            acc, bias + (size_t)l * RR * HD, hb, NN * HD, (l != LL - 1) ? 1 : 0);
    }

    // ---- out = h @ fc_w + fc_b ----
    const int gemm_waves_fc  = (NN / 16) * (OUTS / 64);
    const int gemm_blocks_fc = (gemm_waves_fc * 32 + TB - 1) / TB;
    gemm_bf16_wmma_kernel<<<gemm_blocks_fc, TB, 0, stream>>>(
        hb, Wp_fc, out, fc_b, NN, OUTS);
}
